// VQEmbedding_25099788878015
// MI455X (gfx1250) — matrix operand order repaired and checked by reading, not compiled, not run on hardware
//
#include <hip/hip_runtime.h>
#include <hip/hip_bf16.h>

typedef __attribute__((ext_vector_type(16))) __bf16 v16bf;
typedef __attribute__((ext_vector_type(8)))  __bf16 v8bf;
typedef __attribute__((ext_vector_type(8)))  float  v8f;
typedef __attribute__((ext_vector_type(4)))  unsigned int tdm_v4u;
typedef __attribute__((ext_vector_type(8)))  int          tdm_v8i;
typedef __attribute__((ext_vector_type(4)))  int          tdm_v4i;

#define D_DIM            256
#define K_CODES          8192
#define T_DIM            1024
#define M_ROWS           8192      // B*T
#define NSPLIT           8
#define ROWS_PER_WG      128
#define CODES_PER_SPLIT  (K_CODES / NSPLIT)                   // 1024
#define CODES_PER_STAGE  128
#define STAGES           (CODES_PER_SPLIT / CODES_PER_STAGE)  // 8
#define LDS_STRIDE       264   // bf16 elems per code row: 512B row + 16B pad = 528B (16B aligned, banks advance by 4/row)

#if __has_builtin(__builtin_amdgcn_tensor_load_to_lds) && __has_builtin(__builtin_amdgcn_s_wait_tensorcnt)
#define USE_TDM 1
#else
#define USE_TDM 0
#endif

// ---------------- Kernel 1: e_sq[k] + bf16 codebook copy ----------------
__global__ __launch_bounds__(256) void vq_prep_kernel(const float* __restrict__ cb,
                                                      float* __restrict__ eSq,
                                                      __bf16* __restrict__ cbbf) {
    int k = blockIdx.x * blockDim.x + threadIdx.x;
    const float4* r = (const float4*)(cb + (size_t)k * D_DIM);
    uint2* w = (uint2*)(cbbf + (size_t)k * D_DIM);
    float s = 0.0f;
#pragma unroll 8
    for (int i = 0; i < D_DIM / 4; ++i) {
        float4 v = r[i];
        s += v.x * v.x + v.y * v.y + v.z * v.z + v.w * v.w;
        union { __bf16 bh[4]; uint2 u; } pk;
        pk.bh[0] = (__bf16)v.x; pk.bh[1] = (__bf16)v.y;
        pk.bh[2] = (__bf16)v.z; pk.bh[3] = (__bf16)v.w;
        w[i] = pk.u;
    }
    eSq[k] = s;
}

#if USE_TDM
// Issue one TDM DMA: 128 rows x 256 bf16 (contiguous 64KB tile of bf16 codebook)
// -> LDS with 16B pad after every 512B row (LDS row stride 528B).
__device__ __forceinline__ void tdm_load_tile(const __bf16* gsrc, unsigned ldsOff) {
    unsigned long long ga = (unsigned long long)(uintptr_t)gsrc;
    tdm_v4u g0;
    g0[0] = 1u;                                  // count=1 (valid user descriptor)
    g0[1] = ldsOff;                              // lds_addr (bytes)
    g0[2] = (unsigned)(ga & 0xFFFFFFFFu);        // global_addr[31:0]
    g0[3] = (unsigned)((ga >> 32) & 0x1FFFFFFu)  // global_addr[56:32]
            | (2u << 30);                        // type = 2 ("image")
    tdm_v8i g1;
    g1[0] = (int)((1u << 16)      // data_size = 1 (2 bytes)
                | (1u << 20)      // pad_enable
                | (6u << 22)      // pad_interval: 2^(6+1)=128 dwords (=512B row)
                | (3u << 25));    // pad_amount: 3+1 = 4 dwords (=16B)
    g1[1] = (int)(256u << 16);    // tensor_dim0[15:0]=256 in [31:16]
    g1[2] = (int)(8192u << 16);   // tensor_dim0 hi=0 ; tensor_dim1 lo=8192
    g1[3] = (int)(256u << 16);    // tensor_dim1 hi=0 ; tile_dim0=256
    g1[4] = 128;                  // tile_dim1=128 ; tile_dim2=0
    g1[5] = 256;                  // tensor_dim0_stride lo32 = 256 elems
    g1[6] = 0;                    // stride hi ; tensor_dim1_stride lo (unused, 2D)
    g1[7] = 0;
    tdm_v4i g2 = {0, 0, 0, 0};    // 2D tensor: groups 2/3 unused
    tdm_v4i g3 = {0, 0, 0, 0};
    tdm_v8i g4 = {0, 0, 0, 0, 0, 0, 0, 0};  // clang-23 6-arg form: extra group, unused here
    __builtin_amdgcn_tensor_load_to_lds(g0, g1, g2, g3, g4, 0);
}
#endif

// ---------------- Kernel 2: WMMA distance + per-split argmin ----------------
__global__ __launch_bounds__(256) void vq_argmin_kernel(const float* __restrict__ z,
                                                        const __bf16* __restrict__ cbbf,
                                                        const float* __restrict__ eSq,
                                                        float* __restrict__ pdist,
                                                        int*   __restrict__ pidx) {
    __shared__ __align__(16) __bf16 sB[2][CODES_PER_STAGE * LDS_STRIDE];  // 2 x 67584 B

    const int tid    = threadIdx.x;
    const int wave   = tid >> 5;
    const int lane   = tid & 31;
    const int n_lane = lane & 15;
    const int half   = lane >> 4;

    const int splitBase = blockIdx.y * CODES_PER_SPLIT;

#if USE_TDM
    if (wave == 0) {   // kick off stage 0 DMA before the heavy A-fragment load
        tdm_load_tile(cbbf + (size_t)splitBase * D_DIM,
                      (unsigned)(uintptr_t)&sB[0][0]);
    }
#endif

    const int rowBase = blockIdx.x * ROWS_PER_WG + wave * 16;
    const int myRow   = rowBase + n_lane;
    const int b       = myRow >> 10;
    const int t       = myRow & (T_DIM - 1);
    const float* zcol = z + ((size_t)b << 18) + t;   // element d at zcol[d * 1024]

    // ---- A fragments (16x32 bf16 per k-step, 8 k-steps), ISA 7.12.2 layout ----
    v16bf afrag[8];
#pragma unroll
    for (int s = 0; s < 8; ++s) {
#pragma unroll
        for (int v = 0; v < 8; ++v) {
            int d0 = s * 32 + half * 8 + ((v < 4) ? (2 * v) : (16 + 2 * (v - 4)));
            afrag[s][2 * v]     = (__bf16)zcol[(size_t)d0 * T_DIM];
            afrag[s][2 * v + 1] = (__bf16)zcol[(size_t)(d0 + 1) * T_DIM];
        }
    }

    float minv[8];
    int   mini[8];
#pragma unroll
    for (int v = 0; v < 8; ++v) { minv[v] = 3.4e38f; mini[v] = 0; }

#if USE_TDM
    if (wave == 0) __builtin_amdgcn_s_wait_tensorcnt(0);
    __syncthreads();
#endif

    for (int stage = 0; stage < STAGES; ++stage) {
        const int codeBase = splitBase + stage * CODES_PER_STAGE;
        const int buf = stage & 1;

#if USE_TDM
        if (wave == 0 && stage + 1 < STAGES) {   // DMA next stage while computing this one
            tdm_load_tile(cbbf + (size_t)(codeBase + CODES_PER_STAGE) * D_DIM,
                          (unsigned)(uintptr_t)&sB[buf ^ 1][0]);
        }
        const __bf16* base = &sB[buf][0];
#else
        // Fallback: cooperative bf16 global->LDS copy (512 chunks of 128B, 2/thread)
        {
#pragma unroll
            for (int c2 = 0; c2 < 2; ++c2) {
                int chunk = tid + 256 * c2;          // 0..511
                int code  = chunk >> 2;              // 0..127
                int part  = chunk & 3;               // 128B quarter of the 512B row
                const float4* src = (const float4*)(cbbf + (size_t)(codeBase + code) * D_DIM) + part * 8;
                float4* dst = (float4*)((char*)&sB[0][0] + code * (LDS_STRIDE * 2) + part * 128);
#pragma unroll
                for (int q = 0; q < 8; ++q) dst[q] = src[q];
            }
        }
        __syncthreads();
        const __bf16* base = &sB[0][0];
#endif

#pragma unroll 2
        for (int tile = 0; tile < 8; ++tile) {
            const int nCode = codeBase + tile * 16 + n_lane;
            const __bf16* bcol = base + (tile * 16 + n_lane) * LDS_STRIDE + half * 8;
            v8f acc = {};
            union BU { v16bf v; v8bf h[2]; };
            BU cur, nxt;
            cur.h[0] = *(const v8bf*)(bcol);
            cur.h[1] = *(const v8bf*)(bcol + 16);
#pragma unroll
            for (int s = 0; s < 8; ++s) {
                if (s < 7) {   // prefetch next k-step B fragment during this WMMA
                    nxt.h[0] = *(const v8bf*)(bcol + (s + 1) * 32);
                    nxt.h[1] = *(const v8bf*)(bcol + (s + 1) * 32 + 16);
                }
                acc = __builtin_amdgcn_wmma_f32_16x16x32_bf16(
                        false, afrag[s], false, cur.v, (short)0, acc, false, false);
                if (s < 7) cur = nxt;
            }
            const float eq = eSq[nCode];
#pragma unroll
            for (int v = 0; v < 8; ++v) {
                float dv = eq - 2.0f * acc[v];
                bool better = dv < minv[v];
                mini[v] = better ? nCode : mini[v];
                minv[v] = better ? dv    : minv[v];
            }
        }

#if USE_TDM
        if (wave == 0) __builtin_amdgcn_s_wait_tensorcnt(0);
        __syncthreads();
#else
        __syncthreads();
#endif
    }

    // ---- Cross-lane argmin over the 16 N-lanes (xor masks stay within each half) ----
#pragma unroll
    for (int off = 8; off >= 1; off >>= 1) {
#pragma unroll
        for (int v = 0; v < 8; ++v) {
            float ov = __shfl_xor(minv[v], off, 32);
            int   oi = __shfl_xor(mini[v], off, 32);
            if (ov < minv[v] || (ov == minv[v] && oi < mini[v])) {
                minv[v] = ov; mini[v] = oi;
            }
        }
    }

    if (n_lane == 0) {
        const int split = blockIdx.y;
#pragma unroll
        for (int v = 0; v < 8; ++v) {
            int row = rowBase + v + 8 * half;     // C layout: VGPR v = row v + 8*half
            pdist[(size_t)row * NSPLIT + split] = minv[v];
            pidx [(size_t)row * NSPLIT + split] = mini[v];
        }
    }
}

// ---------------- Kernel 3: reduce NSPLIT partials per row ----------------
__global__ __launch_bounds__(256) void vq_finalize_kernel(const float* __restrict__ pdist,
                                                          const int*   __restrict__ pidx,
                                                          int*         __restrict__ out) {
    int row = blockIdx.x * blockDim.x + threadIdx.x;
    float best = pdist[(size_t)row * NSPLIT];
    int   bi   = pidx [(size_t)row * NSPLIT];
#pragma unroll
    for (int s = 1; s < NSPLIT; ++s) {
        float d = pdist[(size_t)row * NSPLIT + s];
        int   i = pidx [(size_t)row * NSPLIT + s];
        if (d < best || (d == best && i < bi)) { best = d; bi = i; }
    }
    out[row] = bi;
}

extern "C" void kernel_launch(void* const* d_in, const int* in_sizes, int n_in,
                              void* d_out, int out_size, void* d_ws, size_t ws_size,
                              hipStream_t stream) {
    const float* z  = (const float*)d_in[0];   // (8, 256, 1024) f32
    const float* cb = (const float*)d_in[1];   // (8192, 256) f32

    // ws layout: [0, 4MB) bf16 codebook | eSq 32KB | pdist 256KB | pidx 256KB
    __bf16* cbbf = (__bf16*)d_ws;
    size_t  off  = (size_t)K_CODES * D_DIM * 2;
    float* eSq   = (float*)((char*)d_ws + off);            off += (size_t)K_CODES * 4;
    float* pdist = (float*)((char*)d_ws + off);            off += (size_t)M_ROWS * NSPLIT * 4;
    int*   pidx  = (int*)  ((char*)d_ws + off);

    vq_prep_kernel<<<K_CODES / 256, 256, 0, stream>>>(cb, eSq, cbbf);

    dim3 grid(M_ROWS / ROWS_PER_WG, NSPLIT);   // (64, 8)
    vq_argmin_kernel<<<grid, 256, 0, stream>>>(z, cbbf, eSq, pdist, pidx);

    vq_finalize_kernel<<<M_ROWS / 256, 256, 0, stream>>>(pdist, pidx, (int*)d_out);
}
